// OptimizedS4DBlock_8804682957530
// MI455X (gfx1250) — compile-verified, matrix operand order checked by reading, not run on hardware
//
#include <hip/hip_runtime.h>
#include <hip/hip_fp16.h>
#include <hip/hip_bf16.h>

typedef __attribute__((ext_vector_type(16))) _Float16 v16h;
typedef __attribute__((ext_vector_type(8)))  float    v8f;
typedef __attribute__((ext_vector_type(4)))  unsigned int u32x4;
typedef __attribute__((ext_vector_type(8)))  unsigned int u32x8;

#define DMODEL 256
#define NMODE  32        // D_STATE/2 complex modes == wave32 lanes
#define BATCH  16
#define SEQ    4096
#define KDIM   512       // GEMM K = 2*DMODEL (fw|bw concat)
#define NOUT   1024      // 1x1-conv output channels
#define OUTC   512       // GLU / LayerNorm width
#define ROWH   520       // padded LDS row (f16): 1040B = 65*16B -> conflict-free b128

// ---------------------------------------------------------------------------
// Kernel 1: per-(dir,d,mode) complex constants  lam = exp(dt*A),
//           Ce = 2*C*(lam-1)/(A+1e-8)
// ---------------------------------------------------------------------------
__global__ void s4d_consts_kernel(
    const float* __restrict__ log_dt_fw, const float* __restrict__ log_A_real_fw,
    const float* __restrict__ A_imag_fw, const float* __restrict__ C_fw,
    const float* __restrict__ log_dt_bw, const float* __restrict__ log_A_real_bw,
    const float* __restrict__ A_imag_bw, const float* __restrict__ C_bw,
    float* __restrict__ lamRe, float* __restrict__ lamIm,
    float* __restrict__ ceRe,  float* __restrict__ ceIm)
{
    int idx = blockIdx.x * blockDim.x + threadIdx.x;
    if (idx >= 2 * DMODEL * NMODE) return;
    int dir = idx / (DMODEL * NMODE);
    int rem = idx % (DMODEL * NMODE);
    int d = rem / NMODE, n = rem % NMODE;

    const float* ld = dir ? log_dt_bw     : log_dt_fw;
    const float* lA = dir ? log_A_real_bw : log_A_real_fw;
    const float* Ai = dir ? A_imag_bw     : A_imag_fw;
    const float* Cm = dir ? C_bw          : C_fw;

    float dt  = __expf(ld[d]);
    float Are = -__expf(lA[d * NMODE + n]);
    float Aim = Ai[d * NMODE + n];
    float er  = __expf(Are * dt);
    float lre = er * __cosf(Aim * dt);
    float lim = er * __sinf(Aim * dt);

    float dre = Are + 1e-8f, dim = Aim;
    float inv = 1.0f / (dre * dre + dim * dim);
    float nre = lre - 1.0f, nim = lim;
    float rre = (nre * dre + nim * dim) * inv;
    float rim = (nim * dre - nre * dim) * inv;

    float c0 = Cm[(d * NMODE + n) * 2 + 0];
    float c1 = Cm[(d * NMODE + n) * 2 + 1];
    lamRe[idx] = lre;  lamIm[idx] = lim;
    ceRe[idx]  = 2.0f * (c0 * rre - c1 * rim);
    ceIm[idx]  = 2.0f * (c0 * rim + c1 * rre);
}

// ---------------------------------------------------------------------------
// Kernel 2: pack conv_w [1024,512] f32 -> f16 B-fragments (16x16x32 WMMA).
// Fragment layout mirrors ISA 16-bit A striping: lane&15 -> N column,
// lane>=16 selects K halves {8..15,24..31}; v16h comp j: j<8 -> K group 0,
// j>=8 -> K group +16.  One fragment = 32 lanes * 32B contiguous.
// ---------------------------------------------------------------------------
__global__ void s4d_packw_kernel(const float* __restrict__ w,
                                 _Float16* __restrict__ wPack)
{
    int idx = blockIdx.x * blockDim.x + threadIdx.x;   // 1024*512 elements
    int j    = idx & 15;
    int lane = (idx >> 4) & 31;
    int kt   = (idx >> 9) & 15;
    int nt   = idx >> 13;                              // 0..63
    int n = nt * 16 + (lane & 15);
    int k = kt * 32 + (j >> 3) * 16 + (lane >> 4) * 8 + (j & 7);
    wPack[idx] = (_Float16)w[n * KDIM + k];
}

// ---------------------------------------------------------------------------
// Kernel 3: SSM scan. One wave per (b,d,dir); one lane per complex mode.
// Writes gelu(conv + u*Dskip) as f16 into ygelu[B*L, 512].
// ---------------------------------------------------------------------------
__global__ void __launch_bounds__(256) s4d_scan_kernel(
    const float* __restrict__ x, const float* __restrict__ Dskip,
    const float* __restrict__ lamRe, const float* __restrict__ lamIm,
    const float* __restrict__ ceRe,  const float* __restrict__ ceIm,
    _Float16* __restrict__ ygelu)
{
    int lane = threadIdx.x & 31;
    int wid  = (blockIdx.x * blockDim.x + threadIdx.x) >> 5;  // 0..8191
    int d   = wid & (DMODEL - 1);
    int b   = (wid >> 8) & (BATCH - 1);
    int dir = wid >> 12;

    int cbase = dir * DMODEL * NMODE + d * NMODE + lane;
    float lre = lamRe[cbase], lim = lamIm[cbase];
    float cer = ceRe[cbase],  cei = ceIm[cbase];
    float dsk = Dskip[d];

    float sre = 0.0f, sim = 0.0f;
    const float* xb = x + ((size_t)b * SEQ) * DMODEL + d;
    _Float16* yg = ygelu + ((size_t)b * SEQ) * KDIM + dir * DMODEL + d;

    for (int t0 = 0; t0 < SEQ; t0 += 32) {
        int tl  = t0 + lane;
        int ttl = dir ? (SEQ - 1 - tl) : tl;
        float uchunk = xb[(size_t)ttl * DMODEL];   // lane-owned timestep
        float res = 0.0f;
        #pragma unroll
        for (int i = 0; i < 32; ++i) {
            float u = __shfl(uchunk, i, 32);
            float nsre = fmaf(lre, sre, fmaf(-lim, sim, u));
            float nsim = fmaf(lre, sim, lim * sre);
            sre = nsre; sim = nsim;
            float yv = cer * sre - cei * sim;      // Re(Ce * s)
            #pragma unroll
            for (int off = 16; off > 0; off >>= 1)
                yv += __shfl_xor(yv, off, 32);     // all lanes get mode-sum
            if (lane == i) res = yv;               // rotate result to owner
        }
        float v = res + uchunk * dsk;
        float g = 0.5f * v * (1.0f + erff(v * 0.70710678118f));  // exact GELU
        yg[(size_t)ttl * KDIM] = (_Float16)g;      // one vector store / 32 steps
    }
}

// ---------------------------------------------------------------------------
// Kernel 4: fused GEMM (M=65536,K=512,N=1024) + bias + GLU + LayerNorm.
// A tile staged by the Tensor Data Mover (TDM) with native LDS row padding.
// Block tile M=32 x N=1024 (full row -> LN fused). 8 waves: wm in {0,1},
// wn in {0..3}; wave owns cols [wn*128,+128) and the paired +512 set.
// ---------------------------------------------------------------------------
__global__ void __launch_bounds__(256) s4d_gemm_glu_ln_kernel(
    const _Float16* __restrict__ ygelu, const _Float16* __restrict__ wPack,
    const float* __restrict__ bias, const float* __restrict__ gamma,
    const float* __restrict__ beta, float* __restrict__ out)
{
    extern __shared__ __align__(16) char smemRaw[];
    _Float16* As    = (_Float16*)smemRaw;                    // [32][ROWH] @ LDS 0
    float*    part  = (float*)(smemRaw + 32 * ROWH * 2);     // [2][16][4][2]

    int tid = threadIdx.x, lane = tid & 31, wid = tid >> 5;
    int wm = wid >> 2, wn = wid & 3;
    int m0 = blockIdx.x * 32;

    // ---- Stage A tile (32 rows x 1024B) via TDM, 16B pad per row ----------
    // D# group0: count=1, lds_addr=0, global_addr, type=2.
    // D# group1: data_size=3 (8B), pad_enable, pad_interval=7 (256 dw),
    //            pad_amount=3 (4 dw); dims: 128x32 elems, stride 128.
    if (wid == 0) {
        unsigned long long ga =
            (unsigned long long)(uintptr_t)ygelu + (unsigned long long)m0 * (KDIM * 2);
        u32x4 g0;
        g0[0] = 1u;                                   // count=1 (valid user D#)
        g0[1] = 0u;                                   // lds_addr = 0 (As base)
        g0[2] = (unsigned int)ga;                     // global_addr[31:0]
        g0[3] = ((unsigned int)(ga >> 32) & 0x01FFFFFFu) | 0x80000000u; // addr[56:32] | type=2
        u32x8 g1;
        g1[0] = (3u << 16) | (1u << 20) | (7u << 22) | (3u << 25); // dsz=8B,pad 16B/1024B
        g1[1] = (128u << 16);                         // tensor_dim0 = 128 (8B elems)
        g1[2] = (32u << 16);                          // tensor_dim1 = 32 rows
        g1[3] = (128u << 16);                         // tile_dim0 = 128
        g1[4] = 32u;                                  // tile_dim1 = 32
        g1[5] = 128u;                                 // tensor_dim0_stride = 128
        g1[6] = 0u;
        g1[7] = 0u;
        asm volatile("tensor_load_to_lds %0, %1" : : "s"(g0), "s"(g1) : "memory");
        __builtin_amdgcn_s_wait_tensorcnt(0);
    }
    __syncthreads();

    v8f accA[8], accB[8];
    #pragma unroll
    for (int j = 0; j < 8; ++j) { accA[j] = {}; accB[j] = {}; }

    const int aRow  = wm * 16 + (lane & 15);
    const int khalf = (lane >> 4) * 8;

    #pragma unroll 1
    for (int kt = 0; kt < 16; ++kt) {
        // A fragment: ISA 16-bit 16x32 layout -> 2 x ds_load_b128
        v16h aF;
        const _Float16* ap = As + aRow * ROWH + kt * 32 + khalf;
        ((int4*)&aF)[0] = *(const int4*)(ap);        // K {0..7}+khalf
        ((int4*)&aF)[1] = *(const int4*)(ap + 16);   // K {16..23}+khalf
        #pragma unroll
        for (int j = 0; j < 8; ++j) {
            int ntG = wn * 8 + j;                    // first-half columns
            const _Float16* bp = wPack + (((size_t)(ntG * 16 + kt)) * 32 + lane) * 16;
            v16h bF;
            ((int4*)&bF)[0] = *(const int4*)(bp);
            ((int4*)&bF)[1] = *(const int4*)(bp + 8);
            accA[j] = __builtin_amdgcn_wmma_f32_16x16x32_f16(
                false, aF, false, bF, (short)0, accA[j], false, false);

            const _Float16* bq = wPack + (((size_t)((ntG + 32) * 16 + kt)) * 32 + lane) * 16;
            v16h bG;
            ((int4*)&bG)[0] = *(const int4*)(bq);
            ((int4*)&bG)[1] = *(const int4*)(bq + 8);
            accB[j] = __builtin_amdgcn_wmma_f32_16x16x32_f16(
                false, aF, false, bG, (short)0, accB[j], false, false);
        }
    }

    // --- Epilogue: bias + GLU (pairs are in-register), per-row stats ------
    float srow[8] = {}, sq[8] = {};
    #pragma unroll
    for (int j = 0; j < 8; ++j) {
        int col = wn * 128 + j * 16 + (lane & 15);
        float ba = bias[col], bb = bias[col + 512];
        #pragma unroll
        for (int r = 0; r < 8; ++r) {
            float a  = accA[j][r] + ba;
            float bv = accB[j][r] + bb;
            float gv = a * (1.0f / (1.0f + __expf(-bv)));   // GLU
            accA[j][r] = gv;                                // keep for LN
            srow[r] += gv;  sq[r] += gv * gv;
        }
    }
    // Reduce across the 16 lanes of each C-layout row group.
    #pragma unroll
    for (int r = 0; r < 8; ++r) {
        #pragma unroll
        for (int msk = 1; msk < 16; msk <<= 1) {
            srow[r] += __shfl_xor(srow[r], msk, 32);
            sq[r]   += __shfl_xor(sq[r],   msk, 32);
        }
    }
    int half = lane >> 4;   // C layout: reg r -> row r (lanes<16) / r+8 (>=16)
    if ((lane & 15) == 0) {
        #pragma unroll
        for (int r = 0; r < 8; ++r) {
            int R = half * 8 + r;
            part[((wm * 16 + R) * 4 + wn) * 2 + 0] = srow[r];
            part[((wm * 16 + R) * 4 + wn) * 2 + 1] = sq[r];
        }
    }
    __syncthreads();

    float mu[8], rs[8];
    #pragma unroll
    for (int r = 0; r < 8; ++r) {
        int R = half * 8 + r;
        float s = 0.0f, s2 = 0.0f;
        #pragma unroll
        for (int w = 0; w < 4; ++w) {
            s  += part[((wm * 16 + R) * 4 + w) * 2 + 0];
            s2 += part[((wm * 16 + R) * 4 + w) * 2 + 1];
        }
        float m = s * (1.0f / 512.0f);
        float v = s2 * (1.0f / 512.0f) - m * m;
        mu[r] = m;  rs[r] = rsqrtf(v + 1e-5f);
    }

    #pragma unroll
    for (int j = 0; j < 8; ++j) {
        int col = wn * 128 + j * 16 + (lane & 15);
        float gm = gamma[col], bt = beta[col];
        #pragma unroll
        for (int r = 0; r < 8; ++r) {
            int R = half * 8 + r;
            int row = m0 + wm * 16 + R;
            out[(size_t)row * OUTC + col] = (accA[j][r] - mu[r]) * rs[r] * gm + bt;
        }
    }
}

// ---------------------------------------------------------------------------
extern "C" void kernel_launch(void* const* d_in, const int* in_sizes, int n_in,
                              void* d_out, int out_size, void* d_ws, size_t ws_size,
                              hipStream_t stream) {
    const float* x             = (const float*)d_in[0];
    const float* log_dt_fw     = (const float*)d_in[1];
    const float* log_A_real_fw = (const float*)d_in[2];
    const float* A_imag_fw     = (const float*)d_in[3];
    const float* C_fw          = (const float*)d_in[4];
    const float* log_dt_bw     = (const float*)d_in[5];
    const float* log_A_real_bw = (const float*)d_in[6];
    const float* A_imag_bw     = (const float*)d_in[7];
    const float* C_bw          = (const float*)d_in[8];
    const float* Dskip         = (const float*)d_in[9];
    const float* conv_w        = (const float*)d_in[10];
    const float* conv_b        = (const float*)d_in[11];
    const float* ln_gamma      = (const float*)d_in[12];
    const float* ln_beta       = (const float*)d_in[13];
    float* out = (float*)d_out;

    // Workspace: 256KB consts + 1MB packed weights + 64MB f16 activations
    char* ws = (char*)d_ws;
    float* lamRe = (float*)ws;
    float* lamIm = lamRe + 2 * DMODEL * NMODE;
    float* ceRe  = lamIm + 2 * DMODEL * NMODE;
    float* ceIm  = ceRe  + 2 * DMODEL * NMODE;
    _Float16* wPack = (_Float16*)(ceIm + 2 * DMODEL * NMODE);
    _Float16* ygelu = wPack + (size_t)NOUT * KDIM;

    s4d_consts_kernel<<<(2 * DMODEL * NMODE + 255) / 256, 256, 0, stream>>>(
        log_dt_fw, log_A_real_fw, A_imag_fw, C_fw,
        log_dt_bw, log_A_real_bw, A_imag_bw, C_bw,
        lamRe, lamIm, ceRe, ceIm);

    s4d_packw_kernel<<<(NOUT * KDIM) / 256, 256, 0, stream>>>(conv_w, wPack);

    s4d_scan_kernel<<<(2 * BATCH * DMODEL) / 8, 256, 0, stream>>>(
        x, Dskip, lamRe, lamIm, ceRe, ceIm, ygelu);

    size_t smem = (size_t)32 * ROWH * 2 + 2 * 16 * 4 * 2 * sizeof(float);
    s4d_gemm_glu_ln_kernel<<<(BATCH * SEQ) / 32, 256, smem, stream>>>(
        ygelu, wPack, conv_b, ln_gamma, ln_beta, out);
}
